// RegionProposalNetwork_79078937854169
// MI455X (gfx1250) — matrix operand order repaired and checked by reading, not compiled, not run on hardware
//
#include <hip/hip_runtime.h>
#include <math.h>

typedef _Float16 v16h __attribute__((ext_vector_type(16)));
typedef _Float16 v8h  __attribute__((ext_vector_type(8)));
typedef _Float16 v2h  __attribute__((ext_vector_type(2)));
typedef float    v8f  __attribute__((ext_vector_type(8)));

#define HH     50
#define WWID   50
#define NPIX   2500
#define CIN    512
#define CMID   512
#define NBATCH 4
#define NANCH  22500
#define NSORT  32768
#define NPRE   1200
#define NPOST  300
#define IMGSZ  800.0f
#define NMSTH  0.7f
#define MINSZ  16.0f

// d_out regions (floats)
#define OFF_LOCS   0            // 4*22500*4
#define OFF_SCORES 360000       // 4*22500*2
#define OFF_ROIS   540000       // 4*300*4

// workspace byte offsets
#define WSO_W3   ((size_t)0)            // 9*16*32*32*16 f16 = 4,718,592 B
#define WSO_W1   ((size_t)4718592)      // 16*4*32*16 f16    =    65,536 B
#define WSO_H    ((size_t)4784128)      // 4*2500*512 f16 (+pad)
#define WSO_ROI  ((size_t)15040512)     // 4*22500*4 f32
#define WSO_KEY  ((size_t)16480512)     // 4*32768 f32
#define WSO_IDX  ((size_t)17004800)     // 4*32768 i32

__device__ __forceinline__ v16h make_afrag(const _Float16* base) {
  // A frag (16-bit 16x32): this lane's K group0 at base, group1 at base+16
  v8h lo = *(const v8h*)(base);
  v8h hi = *(const v8h*)(base + 16);
  return __builtin_shufflevector(lo, hi, 0,1,2,3,4,5,6,7,8,9,10,11,12,13,14,15);
}

// ---------------- weight swizzle into per-lane B-fragment layout ----------------
// Dense B 32x16 f16: lanes 0-15 hold K=0..7,16..23,
// lanes 16-31 hold K=8..15,24..31; lane = column N. Fragment stored as [lane][16 f16].
__global__ void prep_w3(const float* __restrict__ w1, _Float16* __restrict__ wswz) {
  int i = blockIdx.x * blockDim.x + threadIdx.x;
  if (i >= 9 * 16 * 32 * 512) return;           // 2,359,296
  int e    = i & 15;
  int lane = (i >> 4) & 31;
  int nblk = (i >> 9) & 31;
  int cb   = (i >> 14) & 15;
  int tap  = i >> 18;
  int n  = nblk * 16 + (lane & 15);
  int kc = (e < 8) ? (lane >> 4) * 8 + e : 16 + (lane >> 4) * 8 + (e - 8);
  int c  = cb * 32 + kc;
  wswz[i] = (_Float16)w1[((size_t)n * CIN + c) * 9 + tap];
}

__global__ void prep_w1(const float* __restrict__ wl, const float* __restrict__ wsc,
                        _Float16* __restrict__ wswz1) {
  int i = blockIdx.x * blockDim.x + threadIdx.x;
  if (i >= 16 * 4 * 32 * 16) return;            // 32,768
  int e    = i & 15;
  int lane = (i >> 4) & 31;
  int nblk = (i >> 9) & 3;
  int cb   = i >> 11;
  int n  = nblk * 16 + (lane & 15);
  int kc = (e < 8) ? (lane >> 4) * 8 + e : 16 + (lane >> 4) * 8 + (e - 8);
  int c  = cb * 32 + kc;
  float v = 0.0f;
  if (n < 36)      v = wl[(size_t)n * CMID + c];
  else if (n < 54) v = wsc[(size_t)(n - 36) * CMID + c];
  wswz1[i] = (_Float16)v;
}

// ---------------- 3x3 conv + bias + relu via WMMA implicit GEMM ----------------
// block = 256 thr (8 waves): 32 pixels x 256 out-channels; wave = 16 pix x 64 ch
__global__ __launch_bounds__(256) void conv3_wmma(
    const float* __restrict__ x, const _Float16* __restrict__ wswz,
    const float* __restrict__ b1, _Float16* __restrict__ hbuf) {
  __shared__ _Float16 As[32 * 32];
  const int b      = blockIdx.z;
  const int nbBase = blockIdx.y * 16;   // n-block (16ch) index base
  const int p0     = blockIdx.x * 32;
  const int tid  = threadIdx.x;
  const int lane = tid & 31, wid = tid >> 5;
  const int mi = wid & 1, ni = wid >> 1;
  const int row = lane & 15, half = lane >> 4;

  const v8f vz = {0.f,0.f,0.f,0.f,0.f,0.f,0.f,0.f};
  v8f acc[4] = {vz, vz, vz, vz};

  // A-staging mapping: thread -> (pixel in tile, K pair slice)
  const int sp = tid & 31;
  const int sc = tid >> 5;          // 0..7
  const int pixS = p0 + sp;
  const int ys = pixS / WWID, xs = pixS % WWID;
  const bool pValid = pixS < NPIX;

  for (int tap = 0; tap < 9; ++tap) {
    const int dy = tap / 3 - 1, dx = tap % 3 - 1;
    const int yy = ys + dy, xx = xs + dx;
    const bool inb = pValid && ((unsigned)yy < HH) && ((unsigned)xx < WWID);
    // clamped coords: address always valid -> unconditional load + cndmask zero
    const int yyc = min(max(yy, 0), HH - 1);
    const int xxc = min(max(xx, 0), WWID - 1);
    const size_t xbase = (size_t)b * CIN * NPIX + (size_t)yyc * WWID + xxc;
    for (int cb = 0; cb < 16; ++cb) {
      __syncthreads();
#pragma unroll
      for (int it = 0; it < 2; ++it) {
        int pr = it * 8 + sc;         // K-pair index 0..15
        int kc = pr * 2;
        int c  = cb * 32 + kc;
        float v0 = x[xbase + (size_t)c * NPIX];
        float v1 = x[xbase + (size_t)(c + 1) * NPIX];
        v0 = inb ? v0 : 0.0f;
        v1 = inb ? v1 : 0.0f;
        v2h pk = { (_Float16)v0, (_Float16)v1 };
        *(v2h*)(As + sp * 32 + kc) = pk;     // one b32 LDS store per pair
      }
      __syncthreads();
      v16h af = make_afrag(As + (mi * 16 + row) * 32 + half * 8);
#pragma unroll
      for (int j = 0; j < 4; ++j) {
        int nblk = nbBase + ni * 4 + j;
        const _Float16* bp = wswz + (((size_t)(tap * 16 + cb) * 32 + nblk) * 512) + lane * 16;
        v16h bf = *(const v16h*)bp;
        acc[j] = __builtin_amdgcn_wmma_f32_16x16x32_f16(false, af, false, bf,
                                                        (short)0, acc[j], false, false);
      }
    }
  }
  // epilogue: bias + relu, store h pixel-major f16 (NHWC) for the 1x1 GEMM
#pragma unroll
  for (int j = 0; j < 4; ++j) {
    int ch = (nbBase + ni * 4 + j) * 16 + (lane & 15);
    float bias = b1[ch];
#pragma unroll
    for (int v = 0; v < 8; ++v) {
      int m = v + half * 8;
      int pix = p0 + mi * 16 + m;
      if (pix < NPIX) {
        float val = acc[j][v] + bias;
        val = val > 0.0f ? val : 0.0f;
        hbuf[((size_t)b * NPIX + pix) * CMID + ch] = (_Float16)val;
      }
    }
  }
}

// ---------------- fused 1x1 convs (locs+scores) via WMMA ----------------
__global__ __launch_bounds__(256) void conv1_wmma(
    const _Float16* __restrict__ hbuf, const _Float16* __restrict__ wswz1,
    const float* __restrict__ bl, const float* __restrict__ bs,
    float* __restrict__ dout) {
  const int b = blockIdx.y;
  const int wid = threadIdx.x >> 5, lane = threadIdx.x & 31;
  const int mt = blockIdx.x * 8 + wid;     // 16-pixel tile index (157 tiles)
  if (mt >= 157) return;                    // wave-uniform exit
  const int p0 = mt * 16;
  const int row = lane & 15, half = lane >> 4;
  int pix = p0 + row; if (pix >= NPIX) pix = NPIX - 1;   // clamp loads; store guarded
  const _Float16* hrow = hbuf + ((size_t)b * NPIX + pix) * CMID;

  const v8f vz = {0.f,0.f,0.f,0.f,0.f,0.f,0.f,0.f};
  v8f acc[4] = {vz, vz, vz, vz};
  for (int kc = 0; kc < 16; ++kc) {
    v16h af = make_afrag(hrow + kc * 32 + half * 8);
#pragma unroll
    for (int j = 0; j < 4; ++j) {
      v16h bf = *(const v16h*)(wswz1 + ((size_t)(kc * 4 + j) * 512) + lane * 16);
      acc[j] = __builtin_amdgcn_wmma_f32_16x16x32_f16(false, af, false, bf,
                                                      (short)0, acc[j], false, false);
    }
  }
#pragma unroll
  for (int j = 0; j < 4; ++j) {
    int n = j * 16 + (lane & 15);
#pragma unroll
    for (int v = 0; v < 8; ++v) {
      int pp = p0 + v + half * 8;
      if (pp >= NPIX) continue;
      float val = acc[j][v];
      if (n < 36) {
        dout[OFF_LOCS + (size_t)b * 90000 + (size_t)pp * 36 + n] = val + bl[n];
      } else if (n < 54) {
        int ns = n - 36;
        dout[OFF_SCORES + (size_t)b * 45000 + (size_t)pp * 18 + ns] = val + bs[ns];
      }
    }
  }
}

// ------------- anchors + loc2bbox + clip + filter + softmax-fg -> sort keys -------------
__global__ void prep_sort(const float* __restrict__ dout,
                          float* __restrict__ roi, float* __restrict__ keys,
                          int* __restrict__ idxs) {
  int gi = blockIdx.x * blockDim.x + threadIdx.x;
  if (gi >= NBATCH * NSORT) return;
  int b = gi >> 15;
  int i = gi & (NSORT - 1);
  float key = -__builtin_inff();
  if (i < NANCH) {
    const float* lp = dout + OFF_LOCS + (size_t)b * 90000 + (size_t)i * 4;
    float dyv = lp[0], dxv = lp[1], dh = lp[2], dw = lp[3];
    float s0 = dout[OFF_SCORES + (size_t)b * 45000 + (size_t)i * 2 + 0];
    float s1 = dout[OFF_SCORES + (size_t)b * 45000 + (size_t)i * 2 + 1];
    float mx = fmaxf(s0, s1);
    float e0 = expf(s0 - mx), e1 = expf(s1 - mx);
    float fg = e1 / (e0 + e1);
    int a = i % 9, pixel = i / 9;
    int yI = pixel / WWID, xI = pixel % WWID;
    const float ratios[3] = {0.5f, 1.0f, 2.0f};
    const float scales[3] = {8.0f, 16.0f, 32.0f};
    float r = ratios[a / 3], s = scales[a % 3];
    float hh = 16.0f * s * sqrtf(r);
    float ww = 16.0f * s * sqrtf(1.0f / r);
    float cy = (float)yI * 16.0f + 8.0f;
    float cx = (float)xI * 16.0f + 8.0f;
    float ncy = dyv * hh + cy;
    float ncx = dxv * ww + cx;
    float nh = expf(dh) * hh;
    float nw = expf(dw) * ww;
    float y1 = fminf(fmaxf(ncy - 0.5f * nh, 0.0f), IMGSZ);
    float x1 = fminf(fmaxf(ncx - 0.5f * nw, 0.0f), IMGSZ);
    float y2 = fminf(fmaxf(ncy + 0.5f * nh, 0.0f), IMGSZ);
    float x2 = fminf(fmaxf(ncx + 0.5f * nw, 0.0f), IMGSZ);
    float* rp = roi + ((size_t)b * NANCH + i) * 4;
    rp[0] = y1; rp[1] = x1; rp[2] = y2; rp[3] = x2;
    key = ((y2 - y1) >= MINSZ && (x2 - x1) >= MINSZ) ? fg : -__builtin_inff();
  }
  keys[gi] = key;
  idxs[gi] = i;  // pad idx >= NANCH never wins -inf tie (idx asc tie-break)
}

// ------------- global bitonic sort, key desc / idx asc (matches top_k ties) -------------
__global__ void bitonic_step(float* __restrict__ keys, int* __restrict__ idxs,
                             int j, int k) {
  int b = blockIdx.y;
  int i = blockIdx.x * blockDim.x + threadIdx.x;
  int ixj = i ^ j;
  if (ixj <= i) return;
  float* K = keys + (size_t)b * NSORT;
  int*   I = idxs + (size_t)b * NSORT;
  float ka = K[i], kb = K[ixj];
  int ia = I[i], ib = I[ixj];
  bool desc   = ((i & k) == 0);
  bool aFirst = (ka > kb) || (ka == kb && ia < ib);
  bool doSwap = desc ? (!aFirst) : aFirst;
  if (doSwap) { K[i] = kb; K[ixj] = ka; I[i] = ib; I[ixj] = ia; }
}

// ------------- per-batch sequential NMS over top-1200 + final top-300 select -------------
__global__ __launch_bounds__(1024) void nms_select(
    const float* __restrict__ roi, const float* __restrict__ keys,
    const int* __restrict__ idxs, float* __restrict__ dout) {
  __shared__ float y1[NPRE], x1[NPRE], y2[NPRE], x2[NPRE], area[NPRE], sc[NPRE];
  __shared__ unsigned char sup[NPRE], keep[NPRE];
  __shared__ int sel[NPOST];
  const int b = blockIdx.x, t = threadIdx.x;
  for (int i = t; i < NPRE; i += 1024) {
    int id = idxs[(size_t)b * NSORT + i];
    const float* rp = roi + ((size_t)b * NANCH + id) * 4;
    float a0 = rp[0], a1 = rp[1], a2 = rp[2], a3 = rp[3];
    y1[i] = a0; x1[i] = a1; y2[i] = a2; x2[i] = a3;
    area[i] = (a2 - a0) * (a3 - a1);
    sc[i] = keys[(size_t)b * NSORT + i];
    sup[i] = 0; keep[i] = 0;
  }
  __syncthreads();
  for (int i = 0; i < NPRE; ++i) {
    bool ki = (sup[i] == 0);
    if (t == 0) keep[i] = ki ? 1 : 0;
    if (ki) {
      float iy1 = y1[i], ix1 = x1[i], iy2 = y2[i], ix2 = x2[i], ia = area[i];
      for (int jj = t; jj < NPRE; jj += 1024) {
        if (jj == i) continue;
        float iy = fmaxf(fminf(iy2, y2[jj]) - fmaxf(iy1, y1[jj]), 0.0f);
        float ix = fmaxf(fminf(ix2, x2[jj]) - fmaxf(ix1, x1[jj]), 0.0f);
        float inter = iy * ix;
        float iou = inter / (ia + area[jj] - inter + 1e-9f);
        if (iou > NMSTH) sup[jj] = 1;
      }
    }
    __syncthreads();
  }
  if (t == 0) {
    int cnt = 0;
    for (int i = 0; i < NPRE && cnt < NPOST; ++i)
      if (keep[i] && sc[i] > -3.0e38f) sel[cnt++] = i;
    for (int i = 0; i < NPRE && cnt < NPOST; ++i)
      if (!(keep[i] && sc[i] > -3.0e38f)) sel[cnt++] = i;
  }
  __syncthreads();
  for (int s = t; s < NPOST; s += 1024) {
    int i = sel[s];
    float* op = dout + OFF_ROIS + ((size_t)b * NPOST + s) * 4;
    op[0] = y1[i]; op[1] = x1[i]; op[2] = y2[i]; op[3] = x2[i];
  }
}

extern "C" void kernel_launch(void* const* d_in, const int* in_sizes, int n_in,
                              void* d_out, int out_size, void* d_ws, size_t ws_size,
                              hipStream_t stream) {
  (void)in_sizes; (void)n_in; (void)out_size; (void)ws_size;
  const float* x  = (const float*)d_in[0];
  const float* W1 = (const float*)d_in[1];
  const float* b1 = (const float*)d_in[2];
  const float* Wl = (const float*)d_in[3];
  const float* bl = (const float*)d_in[4];
  const float* Ws = (const float*)d_in[5];
  const float* bs = (const float*)d_in[6];
  float* out = (float*)d_out;
  char*  ws  = (char*)d_ws;

  _Float16* wswz3 = (_Float16*)(ws + WSO_W3);
  _Float16* wswz1 = (_Float16*)(ws + WSO_W1);
  _Float16* hbuf  = (_Float16*)(ws + WSO_H);
  float*    roi   = (float*)(ws + WSO_ROI);
  float*    keys  = (float*)(ws + WSO_KEY);
  int*      idxs  = (int*)(ws + WSO_IDX);

  prep_w3<<<dim3((9 * 16 * 32 * 512) / 256), dim3(256), 0, stream>>>(W1, wswz3);
  prep_w1<<<dim3((16 * 4 * 32 * 16) / 256), dim3(256), 0, stream>>>(Wl, Ws, wswz1);
  conv3_wmma<<<dim3(79, 2, NBATCH), dim3(256), 0, stream>>>(x, wswz3, b1, hbuf);
  conv1_wmma<<<dim3(20, NBATCH), dim3(256), 0, stream>>>(hbuf, wswz1, bl, bs, out);
  prep_sort<<<dim3((NBATCH * NSORT) / 256), dim3(256), 0, stream>>>(out, roi, keys, idxs);
  for (int k = 2; k <= NSORT; k <<= 1)
    for (int j = k >> 1; j >= 1; j >>= 1)
      bitonic_step<<<dim3(NSORT / 256, NBATCH), dim3(256), 0, stream>>>(keys, idxs, j, k);
  nms_select<<<dim3(NBATCH), dim3(1024), 0, stream>>>(roi, keys, idxs, out);
}
